// _InvAttention_73607149519582
// MI455X (gfx1250) — matrix operand order repaired and checked by reading, not compiled, not run on hardware
//
#include <hip/hip_runtime.h>
#include <cstdint>

// ---------------------------------------------------------------------------
// InvAttention for MI455X (gfx1250, wave32, WMMA + Tensor Data Mover).
// Memory-bound problem (~560MB HBM traffic => ~24us floor @ 23.3TB/s).
// bf16 intermediates + v_wmma_f32_16x16x32_bf16; f32 softmax/norms.
// GEMM tiles double-buffered in LDS: bf16 operands DMA'd by TDM
// (tensor_load_to_lds, TENSORcnt) overlapping the WMMA pipe; f32 operands
// staged by VALU (f32->bf16 convert). One barrier per K-step.
// ---------------------------------------------------------------------------

typedef __bf16 bf16_t;
typedef __bf16 v16bf __attribute__((ext_vector_type(16)));
typedef float  v8f   __attribute__((ext_vector_type(8)));
typedef unsigned v4u __attribute__((ext_vector_type(4)));
typedef unsigned v8u __attribute__((ext_vector_type(8)));

constexpr int B_  = 2;
constexpr int NQ  = 2048;
constexpr int NKV = 2048;
constexpr int H_  = 8;
constexpr int D_  = 1024;
constexpr int DH  = 128;

constexpr int BM = 128, BN = 128, BK = 32;   // block tile; 8 waves of 32

// ---- WMMA fragment loaders (ISA 7.12.2 16-bit layouts, wave32) -------------
__device__ inline v16bf frag_a(const bf16_t* lds, int row0, int lane) {
  const int hi = lane >> 4;
  const unsigned* row = (const unsigned*)(lds + (size_t)(row0 + (lane & 15)) * BK);
  union { v16bf v; unsigned u[8]; } f;
#pragma unroll
  for (int p = 0; p < 8; ++p)
    f.u[p] = row[(p & 3) + (hi << 2) + ((p >> 2) << 3)];
  return f.v;
}
__device__ inline v16bf frag_b(const bf16_t* lds, int row0, int lane) {
  const int hi = lane >> 4;
  const unsigned* row = (const unsigned*)(lds + (size_t)(row0 + (lane & 15)) * BK);
  union { v16bf v; unsigned u[8]; } f;
#pragma unroll
  for (int p = 0; p < 8; ++p)
    f.u[p] = row[(p & 3) + (hi << 2) + ((p >> 2) << 3)];
  return f.v;
}

// ---- Tensor Data Mover: DMA a [BM rows x BK k] bf16 tile global->LDS -------
// D# per CDNA5 ISA: 2-group (2D) form, data_size=2B.
// Rows land back-to-back in LDS = the [128][32] layout the frag loaders use.
__device__ inline void tdm_load_tile_bf16(const bf16_t* gsrc, unsigned lds_off,
                                          int Kstride, int rows) {
  const unsigned long long ga = (unsigned long long)(uintptr_t)gsrc;
  v4u g0;
  g0[0] = 1u;                                       // count=1, user descriptor
  g0[1] = lds_off;                                  // lds_addr (bytes)
  g0[2] = (unsigned)(ga & 0xFFFFFFFFu);             // global_addr[31:0]
  g0[3] = (unsigned)((ga >> 32) & 0x1FFFFFFu)       // global_addr[56:32]
          | (2u << 30);                             // type = 2 ("image")
  v8u g1;
  g1[0] = 0x10000u;                                 // data_size=1 -> 2 bytes
  g1[1] = ((unsigned)Kstride & 0xFFFFu) << 16;      // tensor_dim0[15:0]
  g1[2] = (((unsigned)Kstride >> 16) & 0xFFFFu)     // tensor_dim0[31:16]
          | (((unsigned)rows & 0xFFFFu) << 16);     // tensor_dim1[15:0]
  g1[3] = (((unsigned)rows >> 16) & 0xFFFFu)        // tensor_dim1[31:16]
          | ((unsigned)BK << 16);                   // tile_dim0 = 32
  g1[4] = (unsigned)BM;                             // tile_dim1 = 128 (dim2=0)
  g1[5] = (unsigned)Kstride;                        // tensor_dim0_stride[31:0]
  g1[6] = 0u;                                       // stride hi / dim1_stride
  g1[7] = 0u;
  asm volatile("tensor_load_to_lds %0, %1" :: "s"(g0), "s"(g1) : "memory");
}

// ---------------------------------------------------------------------------
// Unified GEMM: C = A[M][K] x W[N][K]^T  (both K-contiguous in memory)
// MODE 0: A=in_w(f32), W=v_inv(f32)  -> head_feats bf16 [b][h][d][k], +in_b[row]
// MODE 1: A=attn(bf16), W=head_feats(bf16), batched z=(b,h), TDM-staged,
//         scaled by weights[b][q][h] -> attn_out bf16 [b][q][e]
// MODE 2: A=attn_out(bf16, TDM), W=out_w(f32) -> d_out f32, +out_b[col]
// ---------------------------------------------------------------------------
template <int MODE>
__global__ __launch_bounds__(256) void gemm_wmma_kernel(
    const void* __restrict__ Ap, const void* __restrict__ Wp,
    const float* __restrict__ bias, const float* __restrict__ weights,
    void* __restrict__ Cp, int M, int N, int K) {
  constexpr bool A_F32 = (MODE == 0);
  constexpr bool W_F32 = (MODE != 1);
  constexpr bool TDM_A = !A_F32;     // bf16 source: stage via TDM
  constexpr bool TDM_W = !W_F32;
  constexpr bool ANY_TDM = TDM_A || TDM_W;

  __shared__ bf16_t ldsA[2][BM * BK];   // 2 x 8 KB (double buffered)
  __shared__ bf16_t ldsB[2][BN * BK];   // 2 x 8 KB (of 320 KB/WGP)

  const int t = threadIdx.x, lane = t & 31, wave = t >> 5;
  const int wm = wave >> 1, wn = wave & 1;          // 4x2 wave grid
  const int n0 = blockIdx.x * BN, m0 = blockIdx.y * BM, z = blockIdx.z;

  const float*  Af  = (const float*)Ap;
  const bf16_t* Abf = (const bf16_t*)Ap;
  const float*  Wf  = (const float*)Wp;
  const bf16_t* Wbf = (const bf16_t*)Wp;
  if (MODE == 1) {                                   // batch = (b,h)
    Abf += (size_t)z * NQ * NKV;
    Wbf += (size_t)z * DH * NKV;
  }

  const unsigned ldsA_off[2] = {(unsigned)(uintptr_t)(void*)&ldsA[0][0],
                                (unsigned)(uintptr_t)(void*)&ldsA[1][0]};
  const unsigned ldsB_off[2] = {(unsigned)(uintptr_t)(void*)&ldsB[0][0],
                                (unsigned)(uintptr_t)(void*)&ldsB[1][0]};

  const int lr = t >> 1;            // 0..127 : tile row this thread stages
  const int lk = (t & 1) << 4;      // 0 or 16: 16 contiguous k elements

  // Stage tile at k-offset kk into buffer pb (TDM issued by wave 0 only;
  // TENSORcnt is per-wave, drained by the issuer before the barrier).
  auto stage = [&](int kk, int pb) {
    if (A_F32) {
      const float* s = Af + (size_t)(m0 + lr) * K + (kk + lk);
#pragma unroll
      for (int i = 0; i < 16; ++i) ldsA[pb][lr * BK + lk + i] = (bf16_t)s[i];
      if (kk + BK < K) __builtin_prefetch(s + BK, 0, 1);
    }
    if (W_F32) {
      const float* s = Wf + (size_t)(n0 + lr) * K + (kk + lk);
#pragma unroll
      for (int i = 0; i < 16; ++i) ldsB[pb][lr * BK + lk + i] = (bf16_t)s[i];
      if (kk + BK < K) __builtin_prefetch(s + BK, 0, 1);
    }
    if (ANY_TDM && wave == 0) {
      if (TDM_A)
        tdm_load_tile_bf16(Abf + (size_t)m0 * K + kk, ldsA_off[pb], K, M);
      if (TDM_W)
        tdm_load_tile_bf16(Wbf + (size_t)n0 * K + kk, ldsB_off[pb], K, N);
    }
  };

  const v8f vzero = {0.f, 0.f, 0.f, 0.f, 0.f, 0.f, 0.f, 0.f};
  v8f acc[2][4];
#pragma unroll
  for (int i = 0; i < 2; ++i)
#pragma unroll
    for (int j = 0; j < 4; ++j) acc[i][j] = vzero;

  // ---- pipelined main loop: load tile t+1 while WMMA-ing tile t ----
  stage(0, 0);
  if (ANY_TDM && wave == 0) __builtin_amdgcn_s_wait_tensorcnt(0);
  __syncthreads();

  int p = 0;
  for (int k0 = 0; k0 < K; k0 += BK) {
    const bool has_next = (k0 + BK < K);
    if (has_next) stage(k0 + BK, p ^ 1);   // DMA overlaps the WMMAs below

    v16bf afr[2], bfr[4];
#pragma unroll
    for (int i = 0; i < 2; ++i)
      afr[i] = frag_a(ldsA[p], wm * 32 + i * 16, lane);
#pragma unroll
    for (int j = 0; j < 4; ++j)
      bfr[j] = frag_b(ldsB[p], wn * 64 + j * 16, lane);
#pragma unroll
    for (int i = 0; i < 2; ++i)
#pragma unroll
      for (int j = 0; j < 4; ++j)
        acc[i][j] = __builtin_amdgcn_wmma_f32_16x16x32_bf16(
            false, afr[i], false, bfr[j], (short)0, acc[i][j], false, false);

    if (has_next) {
      if (ANY_TDM && wave == 0) __builtin_amdgcn_s_wait_tensorcnt(0);
      __syncthreads();                     // publish buffer p^1
      p ^= 1;
    }
  }

  // ---- epilogue (C/D layout: lane n=l&15, VGPR v -> m = v + 8*(l>>4)) ----
#pragma unroll
  for (int i = 0; i < 2; ++i)
#pragma unroll
    for (int j = 0; j < 4; ++j) {
      const int rb = m0 + wm * 32 + i * 16 + (lane >> 4) * 8;
      const int c  = n0 + wn * 64 + j * 16 + (lane & 15);
#pragma unroll
      for (int v = 0; v < 8; ++v) {
        const int r = rb + v;
        const float x = acc[i][j][v];
        if (MODE == 0) {
          // r = e (head feature), c = b*NKV + k ; store [b][h][d][k]
          const int h = r >> 7, d = r & (DH - 1);
          const int b = c >> 11, k = c & (NKV - 1);
          ((bf16_t*)Cp)[((size_t)((b * H_ + h) * DH + d)) * NKV + k] =
              (bf16_t)(x + bias[r]);
        } else if (MODE == 1) {
          // r = q, c = d within head ; scale by L2-norm weight
          const int b = z >> 3, h = z & 7;
          const float w = weights[((size_t)b * NQ + r) * H_ + h];
          ((bf16_t*)Cp)[((size_t)b * NQ + r) * D_ + h * DH + c] =
              (bf16_t)(x * w);
        } else {
          ((float*)Cp)[(size_t)r * N + c] = x + bias[c];
        }
      }
    }
}

// ---------------------------------------------------------------------------
// Masked softmax over Nkv + row L2-norm. One block per (b,q), one wave per
// head: wave32 shuffle reductions; exps parked in LDS (bf16, 32 KB) so the
// normalize pass never re-touches HBM (per-block slice stays in L2 anyway).
// Writes attn bf16 in [b][h][q][k] so stage-3 A rows are k-contiguous.
// ---------------------------------------------------------------------------
__global__ __launch_bounds__(256) void softmax_kernel(
    const float* __restrict__ messages, const int* __restrict__ adj,
    bf16_t* __restrict__ attn, float* __restrict__ weights) {
  __shared__ bf16_t ebuf[H_][NKV];                 // 32 KB
  const int bq = blockIdx.x;
  const int b = bq >> 11, q = bq & (NQ - 1);
  const int h = threadIdx.x >> 5;
  const int lane = threadIdx.x & 31;
  const float* msg = messages + (size_t)bq * NKV * H_;
  const int* am = adj + (size_t)bq * NKV;

  float mx = -3.0e38f;
  for (int k = lane; k < NKV; k += 32) {
    const float x = msg[(size_t)k * H_ + h] + (am[k] ? 0.0f : -1.0e9f);
    mx = fmaxf(mx, x);
  }
#pragma unroll
  for (int off = 16; off; off >>= 1) mx = fmaxf(mx, __shfl_xor(mx, off, 32));

  float s1 = 0.0f, s2 = 0.0f;                      // sum e, sum e^2
  for (int k = lane; k < NKV; k += 32) {
    const float x = msg[(size_t)k * H_ + h] + (am[k] ? 0.0f : -1.0e9f);
    const float e = __expf(x - mx);                // v_exp_f32
    ebuf[h][k] = (bf16_t)e;
    s1 += e; s2 += e * e;
  }
#pragma unroll
  for (int off = 16; off; off >>= 1) {
    s1 += __shfl_xor(s1, off, 32);
    s2 += __shfl_xor(s2, off, 32);
  }
  const float inv = 1.0f / s1;
  if (lane == 0)
    weights[(size_t)bq * H_ + h] = sqrtf(s2) * inv;  // ||p||_2 of softmax row

  bf16_t* dst = attn + (((size_t)(b * H_ + h) * NQ) + q) * NKV;
  for (int k = lane; k < NKV; k += 32)
    dst[k] = (bf16_t)((float)ebuf[h][k] * inv);
}

// ---------------------------------------------------------------------------
// Workspace (needs ~144.2 MB):
//   [0,   8MB)   head_feats bf16 [B][H][DH][NKV]
//   [8, 136MB)   attn       bf16 [B][H][NQ][NKV]
//   [136,144MB)  attn_out   bf16 [B][NQ][D]
//   [144MB, +128KB) weights f32 [B][NQ][H]
// ---------------------------------------------------------------------------
extern "C" void kernel_launch(void* const* d_in, const int* in_sizes, int n_in,
                              void* d_out, int out_size, void* d_ws, size_t ws_size,
                              hipStream_t stream) {
  const float* v_inv    = (const float*)d_in[0];
  const float* messages = (const float*)d_in[1];
  const int*   adj      = (const int*)d_in[2];
  const float* in_w     = (const float*)d_in[3];
  const float* in_b     = (const float*)d_in[4];
  const float* out_w    = (const float*)d_in[5];
  const float* out_b    = (const float*)d_in[6];

  char* ws = (char*)d_ws;
  bf16_t* head_feats = (bf16_t*)(ws);
  bf16_t* attn       = (bf16_t*)(ws + ((size_t)8 << 20));
  bf16_t* attn_out   = (bf16_t*)(ws + ((size_t)136 << 20));
  float*  weights    = (float*)(ws + ((size_t)144 << 20));

  dim3 blk(256);

  // Stage 1: head_feats^T = in_w x v_inv^T  (M=1024 rows e, N=4096 cols b*k)
  gemm_wmma_kernel<0><<<dim3(4096 / BN, 1024 / BM, 1), blk, 0, stream>>>(
      in_w, v_inv, in_b, nullptr, head_feats, 1024, 4096, 1024);

  // Stage 2: masked softmax + L2 weights
  softmax_kernel<<<dim3(B_ * NQ), blk, 0, stream>>>(messages, adj, attn, weights);

  // Stage 3: per-(b,h) attn x head_feats, scaled (M=2048, N=128, K=2048)
  gemm_wmma_kernel<1><<<dim3(1, NQ / BM, B_ * H_), blk, 0, stream>>>(
      attn, head_feats, nullptr, weights, attn_out, NQ, DH, NKV);

  // Stage 4: out = attn_out x out_w^T + out_b (M=4096, N=1024, K=1024)
  gemm_wmma_kernel<2><<<dim3(1024 / BN, 4096 / BM, 1), blk, 0, stream>>>(
      attn_out, out_w, out_b, nullptr, d_out, 4096, 1024, 1024);
}
